// InstanceDiceLoss_56032143343663
// MI455X (gfx1250) — 8-bit matrix operand order repaired and checked by reading, compiled, not run on hardware
//
#include <hip/hip_runtime.h>
#include <hip/hip_bf16.h>


typedef int v8i __attribute__((ext_vector_type(8)));

#define VOXN (128 * 128 * 128)
#define NVOL 4
#define MLAB 65
#define N_OUTER 24
#define K_INNER 8


__device__ __forceinline__ int wave_max_i32(int v) {
#pragma unroll
  for (int o = 16; o > 0; o >>= 1) v = max(v, __shfl_xor(v, o, 32));
  return __builtin_amdgcn_readfirstlane(v);
}

__device__ __forceinline__ unsigned wave_or_u32(unsigned v) {
#pragma unroll
  for (int o = 16; o > 0; o >>= 1) v |= (unsigned)__shfl_xor((int)v, o, 32);
  return (unsigned)__builtin_amdgcn_readfirstlane((int)v);
}

__device__ __forceinline__ unsigned onehot_bytes(unsigned w, unsigned trep) {
  unsigned y  = w ^ trep;
  unsigned nz = (((y & 0x7F7F7F7Fu) + 0x7F7F7F7Fu) | y) & 0x80808080u;
  return (~nz & 0x80808080u) >> 7;
}


__global__ __launch_bounds__(256) void init_labels_kernel(
    const float* __restrict__ src, int* __restrict__ lab) {
  int i = blockIdx.x * 256 + threadIdx.x;
  lab[i] = (src[i] > 0.5f) ? (i + 1) : 0;
}


__device__ __forceinline__ int nmax27(const int* t, int idx) {
  int m = 0;
#pragma unroll
  for (int dz = -1; dz <= 1; ++dz)
#pragma unroll
    for (int dy = -1; dy <= 1; ++dy)
#pragma unroll
      for (int dx = -1; dx <= 1; ++dx)
        m = max(m, t[idx + dz * 100 + dy * 10 + dx]);
  return m;
}

__global__ __launch_bounds__(256) void propagate_kernel(
    const int* __restrict__ src, int* __restrict__ dst) {
  __shared__ int T[1000];
  const int bx = blockIdx.x * 8, by = blockIdx.y * 8, bz = blockIdx.z * 8;
  const int tid = threadIdx.x + threadIdx.y * 8 + threadIdx.z * 64;

  for (int l = tid; l < 1000; l += 256) {
    int lz = l / 100, r = l % 100, ly = r / 10, lx = r % 10;
    int gx = bx + lx - 1, gy = by + ly - 1, gz = bz + lz - 1;
    int v = 0;
    if ((unsigned)gx < 128u && (unsigned)gy < 128u && (unsigned)gz < 128u)
      v = src[(gz * 128 + gy) * 128 + gx];
    T[l] = v;
  }
  __syncthreads();

  const int x = threadIdx.x + 1, y = threadIdx.y + 1;
  const int i0 = ((threadIdx.z + 1) * 10 + y) * 10 + x;
  const int i1 = ((threadIdx.z + 5) * 10 + y) * 10 + x;

#pragma unroll 1
  for (int s = 0; s < K_INNER; ++s) {
    int c0 = T[i0], c1 = T[i1];
    int n0 = (c0 > 0) ? nmax27(T, i0) : 0;
    int n1 = (c1 > 0) ? nmax27(T, i1) : 0;
    __syncthreads();
    T[i0] = n0;
    T[i1] = n1;
    __syncthreads();
  }

  dst[((bz + threadIdx.z) * 128 + by + threadIdx.y) * 128 + bx + threadIdx.x] = T[i0];
  dst[((bz + threadIdx.z + 4) * 128 + by + threadIdx.y) * 128 + bx + threadIdx.x] = T[i1];
}


__global__ void zero_reps_kernel(int* reps) {
  int t = threadIdx.x;
  reps[t] = (t == 0) ? 0 : 0x7FFFFFFF;
}

__global__ __launch_bounds__(256) void find_reps_kernel(
    const int* __restrict__ lab, int* __restrict__ reps) {
  int i = blockIdx.x * 256 + threadIdx.x;
  int l = lab[i];
  if (l == i + 1 && l != 0) {
    int pos = atomicAdd(&reps[0], 1);
    if (pos < 127) reps[1 + pos] = l;
  }
}

__global__ __launch_bounds__(128) void sort_reps_kernel(int* reps) {
  __shared__ int s[128];
  int t = threadIdx.x;
  s[t] = (t < 127) ? reps[1 + t] : 0x7FFFFFFF;
  for (int k = 2; k <= 128; k <<= 1) {
    for (int j = k >> 1; j > 0; j >>= 1) {
      __syncthreads();
      int ix = t ^ j;
      int a = s[t], b = s[ix];
      bool up = ((t & k) == 0);
      int r = (t < ix) ? (up ? min(a, b) : max(a, b))
                       : (up ? max(a, b) : min(a, b));
      __syncthreads();
      s[t] = r;
    }
  }
  __syncthreads();
  if (t < 127) reps[1 + t] = s[t];
}

__global__ __launch_bounds__(256) void compact_kernel(
    const int* __restrict__ lab, const int* __restrict__ reps,
    unsigned char* __restrict__ cmp) {
  int i = blockIdx.x * 256 + threadIdx.x;
  int l = lab[i];
  unsigned char id = 0;
  if (l > 0) {
    int cnt = min(reps[0], 127);
    int lo = 0, hi = cnt;
    while (lo < hi) {
      int mid = (lo + hi) >> 1;
      if (reps[1 + mid] < l) lo = mid + 1; else hi = mid;
    }
    id = (unsigned char)min(lo + 1, 65);
  }
  cmp[i] = id;
}


__global__ void zero_overlap_kernel(unsigned* O) {
  for (int i = threadIdx.x; i < MLAB * MLAB; i += 256) O[i] = 0;
}

__global__ __launch_bounds__(256) void overlap_kernel(
    const unsigned char* __restrict__ gt, const unsigned char* __restrict__ pr,
    unsigned* __restrict__ O) {
  __shared__ unsigned hist[MLAB * MLAB];
  __shared__ unsigned stage[8][32];

  const int tid = threadIdx.x;
  const int lane = tid & 31;
  const int w = tid >> 5;
  for (int i = tid; i < MLAB * MLAB; i += 256) hist[i] = 0;
  __syncthreads();

  const int waveG = (blockIdx.x << 3) + w;
  const int rowSel = lane >> 4;
  const int sub = lane & 15;
  unsigned bg = 0;

  for (int chunk = waveG; chunk < (2 * VOXN) / 64; chunk += 512) {
    size_t base = (size_t)chunk * 64;
    __builtin_prefetch(gt + base + 64 * 512, 0, 1);
    __builtin_prefetch(pr + base + 64 * 512, 0, 1);
    unsigned gs = ((const unsigned short*)(gt + base))[lane];
    unsigned ps = ((const unsigned short*)(pr + base))[lane];
    unsigned g0 = gs & 255u, g1 = gs >> 8, p0 = ps & 255u, p1 = ps >> 8;

    if (wave_max_i32((int)(g0 | g1 | p0 | p1)) == 0) {
      bg += 2;
      continue;
    }

    ((unsigned short*)&stage[w][0])[lane]  = (unsigned short)gs;
    ((unsigned short*)&stage[w][16])[lane] = (unsigned short)ps;
    asm volatile("s_wait_dscnt 0" ::: "memory");

    unsigned gm = wave_or_u32((1u << (g0 >> 4)) | (1u << (g1 >> 4)));
    unsigned pm = wave_or_u32((1u << (p0 >> 4)) | (1u << (p1 >> 4)));
    const unsigned* sg = &stage[w][0];
    const unsigned* sp = &stage[w][16];

    for (int gB = 0; gB < 5; ++gB) {
      if (!((gm >> gB) & 1)) continue;
      v8i A;
      unsigned tg = (unsigned)(gB * 16 + sub) * 0x01010101u;
#pragma unroll
      for (int r = 0; r < 8; ++r) {
        int d = ((r >> 1) << 2) + (r & 1) + (rowSel << 1);
        A[r] = (int)onehot_bytes(sg[d], tg);
      }
      for (int pB = 0; pB < 5; ++pB) {
        if (!((pm >> pB) & 1)) continue;
        v8i B;
        unsigned tp = (unsigned)(pB * 16 + sub) * 0x01010101u;
#pragma unroll
        for (int r = 0; r < 8; ++r) {
          int d = ((r >> 1) << 2) + (r & 1) + (rowSel << 1);
          B[r] = (int)onehot_bytes(sp[d], tp);
        }
        v8i C = {};
        v8i D = __builtin_amdgcn_wmma_i32_16x16x64_iu8(
            false, A, false, B, C, false, false);
#pragma unroll
        for (int r = 0; r < 8; ++r) {
          int val = D[r];
          int row = gB * 16 + r + (rowSel << 3);
          int col = pB * 16 + sub;
          if (val != 0 && row < MLAB && col < MLAB)
            atomicAdd(&hist[row * MLAB + col], (unsigned)val);
        }
      }
    }
  }

  atomicAdd(&hist[0], bg);
  __syncthreads();
  for (int i = tid; i < MLAB * MLAB; i += 256) {
    unsigned v = hist[i];
    if (v) atomicAdd(&O[i], v);
  }
}


__global__ __launch_bounds__(128) void finalize_kernel(
    const unsigned* __restrict__ O, float* __restrict__ out) {
  __shared__ float gt_size[MLAB], pr_size[MLAB], dice[MLAB];
  __shared__ int gtHas[MLAB], fpFlag[MLAB];
  int t = threadIdx.x;
  if (t < MLAB) {
    float rs = 0.f, cs = 0.f;
    for (int p = 0; p < MLAB; ++p) rs += (float)O[t * MLAB + p];
    for (int g = 0; g < MLAB; ++g) cs += (float)O[g * MLAB + t];
    gt_size[t] = rs;
    pr_size[t] = cs;
  }
  __syncthreads();
  if (t >= 1 && t < MLAB) {
    float inter = 0.f, uni = 0.f;
    for (int p = 1; p < MLAB; ++p) {
      float v = (float)O[t * MLAB + p];
      inter += v;
      if (v > 0.f) uni += pr_size[p];
    }
    dice[t] = (inter > 0.f) ? (2.f * inter / fmaxf(uni + gt_size[t], 1.f)) : 0.f;
    gtHas[t] = (gt_size[t] > 0.f) ? 1 : 0;
    bool tp = false;
    for (int g = 1; g < MLAB; ++g)
      if (O[g * MLAB + t] > 0) { tp = true; break; }
    fpFlag[t] = (pr_size[t] > 0.f && !tp) ? 1 : 0;
  }
  __syncthreads();
  if (t == 0) {
    float ld = 0.f;
    int ng = 0, nf = 0;
    for (int g = 1; g < MLAB; ++g) { ld += dice[g]; ng += gtHas[g]; nf += fpFlag[g]; }
    out[0] = ld / (float)(ng + nf);
  }
}


extern "C" void kernel_launch(void* const* d_in, const int* in_sizes, int n_in,
                              void* d_out, int out_size, void* d_ws, size_t ws_size,
                              hipStream_t stream) {
  (void)in_sizes; (void)n_in; (void)out_size; (void)ws_size;
  const float* x = (const float*)d_in[0];
  const float* y = (const float*)d_in[1];
  float* out = (float*)d_out;

  char* ws = (char*)d_ws;
  int* labA = (int*)(ws);
  int* labB = (int*)(ws + (size_t)(8u << 20));
  unsigned char* cmp = (unsigned char*)(ws + (size_t)(16u << 20));
  int* reps = (int*)(ws + (size_t)(24u << 20));
  unsigned* overlap = (unsigned*)(ws + (size_t)(24u << 20) + 1024);

  const float* srcs[NVOL] = { y, y + VOXN, x, x + VOXN };

  for (int v = 0; v < NVOL; ++v) {
    init_labels_kernel<<<VOXN / 256, 256, 0, stream>>>(srcs[v], labA);
    for (int it = 0; it < N_OUTER; ++it) {
      propagate_kernel<<<dim3(16, 16, 16), dim3(8, 8, 4), 0, stream>>>(labA, labB);
      int* t = labA; labA = labB; labB = t;
    }
    zero_reps_kernel<<<1, 128, 0, stream>>>(reps);
    find_reps_kernel<<<VOXN / 256, 256, 0, stream>>>(labA, reps);
    sort_reps_kernel<<<1, 128, 0, stream>>>(reps);
    compact_kernel<<<VOXN / 256, 256, 0, stream>>>(labA, reps, cmp + (size_t)v * VOXN);
  }

  zero_overlap_kernel<<<1, 256, 0, stream>>>(overlap);
  overlap_kernel<<<64, 256, 0, stream>>>(cmp  ,
                                         cmp + (size_t)2 * VOXN  ,
                                         overlap);
  finalize_kernel<<<1, 128, 0, stream>>>(overlap, out);
}
